// CFTransform_51960514347422
// MI455X (gfx1250) — matrix operand order repaired and checked by reading, not compiled, not run on hardware
//
#include <hip/hip_runtime.h>
#include <hip/hip_bf16.h>

// ---------------------------------------------------------------------------
// CDNA5 (gfx1250) WMMA types and helpers
// ---------------------------------------------------------------------------
typedef __attribute__((ext_vector_type(16))) __bf16 v16bf;
typedef __attribute__((ext_vector_type(8)))  float  v8f;

union Frag {
    v16bf          v;
    unsigned int   d[8];
    unsigned short u[16];
};

__device__ __forceinline__ unsigned short f2bf(float f) {
    unsigned int x = __float_as_uint(f);
    x += 0x7FFFu + ((x >> 16) & 1u);       // round-to-nearest-even
    return (unsigned short)(x >> 16);
}

// ---------------------------------------------------------------------------
// Weight transpose + bf16 convert:  dst[k*Npad + n] = src[n*K + k] (0-pad n>=Nsrc)
// ---------------------------------------------------------------------------
__global__ void cft_wtrans(const float* __restrict__ src, unsigned short* __restrict__ dst,
                           int K, int Nsrc, int Npad, long long sSrc, long long sDst) {
    int batch = blockIdx.y;
    int t = blockIdx.x * 256 + threadIdx.x;
    if (t >= K * Npad) return;
    int n = t % Npad, k = t / Npad;
    float v = (n < Nsrc) ? src[batch * sSrc + (size_t)n * K + k] : 0.0f;
    dst[batch * sDst + (size_t)k * Npad + n] = (n < Nsrc) ? f2bf(v) : (unsigned short)0;
}

// ---------------------------------------------------------------------------
// Depthwise 3x3 conv (SAME) + bias, NCHW in -> [b*HW, C] bf16 out
// ---------------------------------------------------------------------------
__global__ void cft_dwconv3x3(const float* __restrict__ x, const float* __restrict__ w,
                              const float* __restrict__ bias, unsigned short* __restrict__ y,
                              int Bn, int C, int H, int Wd) {
    size_t t = (size_t)blockIdx.x * 256 + threadIdx.x;
    size_t total = (size_t)Bn * C * H * Wd;
    if (t >= total) return;
    int wd = (int)(t % Wd); size_t r = t / Wd;
    int h = (int)(r % H);   r /= H;
    int c = (int)(r % C);
    int b = (int)(r / C);
    const float* xp = x + ((size_t)(b * C + c)) * H * Wd;
    const float* wp = w + c * 9;
    float s = bias[c];
#pragma unroll
    for (int kh = -1; kh <= 1; ++kh) {
        int hh = h + kh;
        if (hh < 0 || hh >= H) continue;
#pragma unroll
        for (int kw = -1; kw <= 1; ++kw) {
            int ww = wd + kw;
            if (ww < 0 || ww >= Wd) continue;
            s += xp[hh * Wd + ww] * wp[(kh + 1) * 3 + (kw + 1)];
        }
    }
    y[((size_t)b * H * Wd + h * Wd + wd) * C + c] = f2bf(s);
}

// ---------------------------------------------------------------------------
// NCHW fp32 -> [b, p, c] bf16
// ---------------------------------------------------------------------------
__global__ void cft_nchw_to_pc(const float* __restrict__ x, unsigned short* __restrict__ y,
                               int Bn, int C, int HW) {
    size_t t = (size_t)blockIdx.x * 256 + threadIdx.x;
    size_t total = (size_t)Bn * C * HW;
    if (t >= total) return;
    int p = (int)(t % HW); size_t r = t / HW;
    int c = (int)(r % C);
    int b = (int)(r / C);
    y[((size_t)b * HW + p) * C + c] = f2bf(x[t]);
}

// ---------------------------------------------------------------------------
// Tiled bf16 WMMA GEMM.  C[M,N] = (A[M,K] * B[K,N] + bias) * alpha
// Block = 256 threads (8 waves), tile 64(M) x 128(N), K-step 32.
// Branchless staging: out-of-range A rows are clamped (row m only feeds output
// row m; masked in epilogue).  B requires ldb == stored row width; the 16-col
// chunk start is clamped to ldb-16, so edge-tile reads stay in-bounds and the
// mis-positioned data only lands in output columns >= nValid (never stored).
// Batched: z = outer*innerCount + inner with independent strides.
// ---------------------------------------------------------------------------
__global__ __launch_bounds__(256) void cft_gemm_bf16(
    const unsigned short* __restrict__ A, int lda, long long sAo, long long sAi,
    const unsigned short* __restrict__ Bm, int ldb, long long sBo, long long sBi,
    float* __restrict__ Cf, int ldc, long long sCo, long long sCi, int cColMajor,
    unsigned short* __restrict__ Cb, int ldcb, long long sCbo, long long sCbi,
    const float* __restrict__ bias, float alpha,
    int M, int N, int K, int innerCount, int nValid) {

    __shared__ unsigned int AsU[64 * 20];   // 64 rows x 32 bf16 (pad to 40)
    __shared__ unsigned int BsU[32 * 68];   // 32 rows x 128 bf16 (pad to 136)

    const int tid  = threadIdx.x;
    const int wave = tid >> 5, lane = tid & 31;
    const int hf = lane >> 4, l15 = lane & 15;
    const int waveM = wave >> 2, waveN = wave & 3;
    const int tileM = blockIdx.y * 64, tileN = blockIdx.x * 128;
    const int z = blockIdx.z;
    const int inner = z % innerCount, outer = z / innerCount;

    const unsigned short* Ab = A + outer * sAo + inner * sAi;
    const unsigned short* Bb = Bm + outer * sBo + inner * sBi;

    v8f acc[2][2];
#pragma unroll
    for (int i = 0; i < 2; ++i)
#pragma unroll
        for (int j = 0; j < 2; ++j) acc[i][j] = (v8f)0.0f;

    const int aRow = tid >> 2;          // 0..63
    const int aK   = (tid & 3) * 8;     // 0,8,16,24
    const int bK   = tid >> 3;          // 0..31
    const int bN   = (tid & 7) * 16;    // 0..112

    // clamped (branchless) staging coordinates
    int gM = tileM + aRow;  if (gM >= M) gM = M - 1;
    int gN0 = tileN + bN;   if (gN0 + 16 > ldb) gN0 = ldb - 16;
    const unsigned short* Ap = Ab + (size_t)gM * lda + aK;
    const unsigned short* Bp = Bb + (size_t)bK * ldb + gN0;

    const int nK = K >> 5;
    for (int kc = 0; kc < nK; ++kc) {
        // ---- stage A tile (one 16B load per thread) ----
        uint4 av = *(const uint4*)(Ap + kc * 32);
        *(uint4*)&AsU[aRow * 20 + (aK >> 1)] = av;

        // ---- stage B tile (two 16B loads per thread) ----
        const unsigned short* p = Bp + (size_t)(kc * 32) * ldb;
        uint4 b0 = *(const uint4*)p;
        uint4 b1 = *(const uint4*)(p + 8);
        *(uint4*)&BsU[bK * 68 + (bN >> 1)]     = b0;
        *(uint4*)&BsU[bK * 68 + (bN >> 1) + 4] = b1;

        __syncthreads();

        // prefetch next K-slice into GL2 while we compute
        if (kc + 1 < nK) {
            __builtin_prefetch(Ap + (kc + 1) * 32, 0, 0);
            __builtin_prefetch(Bp + (size_t)((kc + 1) * 32) * ldb, 0, 0);
        }

        // ---- fragments (ISA 16-bit A 16x32 and B 32x16 layouts) ----
        Frag a[2], bb[2];
#pragma unroll
        for (int im = 0; im < 2; ++im) {
            int row = waveM * 32 + im * 16 + l15;
#pragma unroll
            for (int j = 0; j < 8; ++j)
                a[im].d[j] = AsU[row * 20 + ((j < 4) ? j : j + 4) + hf * 4];
        }
#pragma unroll
        for (int in = 0; in < 2; ++in) {
            const unsigned short* Bs16 = (const unsigned short*)BsU;
#pragma unroll
            for (int s = 0; s < 16; ++s)
                bb[in].u[s] = Bs16[((s & 7) + hf * 8 + (s >> 3) * 16) * 136 + waveN * 32 + in * 16 + l15];
        }
#pragma unroll
        for (int im = 0; im < 2; ++im)
#pragma unroll
            for (int in = 0; in < 2; ++in)
                acc[im][in] = __builtin_amdgcn_wmma_f32_16x16x32_bf16(
                    false, a[im].v, false, bb[in].v, (short)0, acc[im][in], false, false);

        __syncthreads();
    }

    // ---- epilogue ----
    float* Cfb = Cf ? Cf + outer * sCo + inner * sCi : nullptr;
    unsigned short* Cbb = Cb ? Cb + outer * sCbo + inner * sCbi : nullptr;
#pragma unroll
    for (int im = 0; im < 2; ++im)
#pragma unroll
        for (int in = 0; in < 2; ++in)
#pragma unroll
            for (int j = 0; j < 8; ++j) {
                int gR = tileM + waveM * 32 + im * 16 + hf * 8 + j;
                int gC = tileN + waveN * 32 + in * 16 + l15;
                if (gR < M && gC < nValid) {
                    float val = acc[im][in][j];
                    if (bias) val += bias[gC];
                    val *= alpha;
                    if (Cfb)
                        Cfb[cColMajor ? ((size_t)gC * ldc + gR)
                                      : ((size_t)gR * ldc + gC)] = val;
                    if (Cbb) Cbb[(size_t)gR * ldcb + gC] = f2bf(val);
                }
            }
}

// ---------------------------------------------------------------------------
// Softmax over spatial (HW) per (b, n); fp32 in -> bf16 out ([B,160,HW], rows
// n>=150 pre-zeroed by memset).
// ---------------------------------------------------------------------------
__global__ __launch_bounds__(256) void cft_softmax_hw(
    const float* __restrict__ mm, unsigned short* __restrict__ mask, int HW) {
    const int n = blockIdx.x, b = blockIdx.y;
    const float* x = mm + ((size_t)b * 150 + n) * HW;
    unsigned short* y = mask + ((size_t)b * 160 + n) * HW;
    __shared__ float red[256];
    const int tid = threadIdx.x;

    float mx = -3.4e38f;
    for (int i = tid; i < HW; i += 256) mx = fmaxf(mx, x[i]);
    red[tid] = mx; __syncthreads();
    for (int st = 128; st > 0; st >>= 1) {
        if (tid < st) red[tid] = fmaxf(red[tid], red[tid + st]);
        __syncthreads();
    }
    mx = red[0]; __syncthreads();

    float sm = 0.0f;
    for (int i = tid; i < HW; i += 256) sm += __expf(x[i] - mx);
    red[tid] = sm; __syncthreads();
    for (int st = 128; st > 0; st >>= 1) {
        if (tid < st) red[tid] += red[tid + st];
        __syncthreads();
    }
    sm = red[0]; __syncthreads();

    float r = 1.0f / sm;
    for (int i = tid; i < HW; i += 256) y[i] = f2bf(__expf(x[i] - mx) * r);
}

// ---------------------------------------------------------------------------
// Repack kv fp32 [B*160, 1024] -> kT bf16 [B,8,64,160] and v bf16 [B,8,160,64]
// (n >= 150 zero-padded)
// ---------------------------------------------------------------------------
__global__ void cft_repack_kv(const float* __restrict__ kvf,
                              unsigned short* __restrict__ kT,
                              unsigned short* __restrict__ vB, int Bn) {
    size_t t = (size_t)blockIdx.x * 256 + threadIdx.x;
    size_t total = (size_t)Bn * 160 * 1024;
    if (t >= total) return;
    int co = (int)(t % 1024); size_t r = t / 1024;
    int n = (int)(r % 160);
    int b = (int)(r / 160);
    float val = (n < 150) ? kvf[t] : 0.0f;
    unsigned short bv = (n < 150) ? f2bf(val) : (unsigned short)0;
    if (co < 512) {
        int h = co >> 6, d = co & 63;
        kT[(((size_t)(b * 8 + h) * 64) + d) * 160 + n] = bv;
    } else {
        int c2 = co - 512, h = c2 >> 6, d = c2 & 63;
        vB[(((size_t)(b * 8 + h) * 160) + n) * 64 + d] = bv;
    }
}

// ---------------------------------------------------------------------------
// Fused attention: per (b, h, 64 q-rows).  4 waves, each wave owns 16 rows.
// logits = q(16x64) x kT(64x160) via WMMA; inverted-max softmax
// (softmax(max-x) == exp(min-x)/sum); attn_mean atomically accumulated;
// out = P(16x160) x V(160x64) via WMMA.
// ---------------------------------------------------------------------------
__global__ __launch_bounds__(128) void cft_attention(
    const unsigned short* __restrict__ q,   // [B*HW, 512] (already *0.125)
    const unsigned short* __restrict__ kT,  // [B,8,64,160]
    const unsigned short* __restrict__ vB,  // [B,8,160,64]
    unsigned short* __restrict__ outb,      // [B*HW, 512]
    float* __restrict__ attn_mean,          // [B,HW,150]
    int HW) {

    __shared__ unsigned int kU[64 * 84];    // 64 x 160 bf16 (pad 168)
    __shared__ unsigned int vU[160 * 36];   // 160 x 64 bf16 (pad 72)
    __shared__ unsigned int pU[4 * 16 * 84];// per-wave 16 x 160 bf16 (pad 168)

    const int tid = threadIdx.x, wave = tid >> 5, lane = tid & 31;
    const int hf = lane >> 4, l15 = lane & 15;
    const int p0 = blockIdx.x * 64 + wave * 16;
    const int h = blockIdx.y, b = blockIdx.z;

    const unsigned int* kg = (const unsigned int*)(kT + ((size_t)(b * 8 + h) * 64) * 160);
    const unsigned int* vg = (const unsigned int*)(vB + ((size_t)(b * 8 + h) * 160) * 64);
    for (int idx = tid; idx < 64 * 80;  idx += 128) kU[(idx / 80) * 84 + (idx % 80)] = kg[idx];
    for (int idx = tid; idx < 160 * 32; idx += 128) vU[(idx / 32) * 36 + (idx % 32)] = vg[idx];
    __syncthreads();

    // q A-fragments straight from global (row = p0 + l15, 64 cols in 2 chunks)
    const unsigned int* qrow =
        (const unsigned int*)q + (((size_t)(b * HW + p0 + l15)) * 512 + (size_t)h * 64) / 2;
    Frag aq[2];
#pragma unroll
    for (int kc = 0; kc < 2; ++kc)
#pragma unroll
        for (int j = 0; j < 8; ++j)
            aq[kc].d[j] = qrow[kc * 16 + ((j < 4) ? j : j + 4) + hf * 4];

    // logits: 10 N-tiles x 2 K-chunks
    v8f s[10];
#pragma unroll
    for (int nt = 0; nt < 10; ++nt) s[nt] = (v8f)0.0f;
#pragma unroll
    for (int nt = 0; nt < 10; ++nt) {
#pragma unroll
        for (int kc = 0; kc < 2; ++kc) {
            Frag bk;
            const unsigned short* k16 = (const unsigned short*)kU;
#pragma unroll
            for (int s = 0; s < 16; ++s) bk.u[s] = k16[((s & 7) + hf * 8 + (s >> 3) * 16 + kc * 32) * 168 + nt * 16 + l15];
            s[nt] = __builtin_amdgcn_wmma_f32_16x16x32_bf16(
                false, aq[kc].v, false, bk.v, (short)0, s[nt], false, false);
        }
    }

    // ---- softmax(max - x) == exp(min - x) / sum (valid cols only) ----
    float mn[8];
#pragma unroll
    for (int j = 0; j < 8; ++j) mn[j] = 3.4e38f;
#pragma unroll
    for (int nt = 0; nt < 10; ++nt) {
        int c = nt * 16 + l15;
        if (c < 150) {
#pragma unroll
            for (int j = 0; j < 8; ++j) mn[j] = fminf(mn[j], s[nt][j]);
        }
    }
#pragma unroll
    for (int m = 1; m < 16; m <<= 1)
#pragma unroll
        for (int j = 0; j < 8; ++j) mn[j] = fminf(mn[j], __shfl_xor(mn[j], m, 32));

    float sum[8];
#pragma unroll
    for (int j = 0; j < 8; ++j) sum[j] = 0.0f;
#pragma unroll
    for (int nt = 0; nt < 10; ++nt) {
        int c = nt * 16 + l15;
        bool valid = c < 150;
#pragma unroll
        for (int j = 0; j < 8; ++j) {
            float pv = valid ? __expf(mn[j] - s[nt][j]) : 0.0f;
            s[nt][j] = pv;
            sum[j] += pv;
        }
    }
#pragma unroll
    for (int m = 1; m < 16; m <<= 1)
#pragma unroll
        for (int j = 0; j < 8; ++j) sum[j] += __shfl_xor(sum[j], m, 32);
    float rs[8];
#pragma unroll
    for (int j = 0; j < 8; ++j) rs[j] = 1.0f / sum[j];

    // write probabilities: attn_mean (+= p/8) and bf16 staging in LDS
    unsigned short* ps = (unsigned short*)pU;
#pragma unroll
    for (int nt = 0; nt < 10; ++nt) {
        int c = nt * 16 + l15;
#pragma unroll
        for (int j = 0; j < 8; ++j) {
            float pv = s[nt][j] * rs[j];
            int pr = p0 + hf * 8 + j;
            if (c < 150)
                atomicAdd(attn_mean + ((size_t)b * HW + pr) * 150 + c, pv * 0.125f);
            ps[(size_t)(wave * 16 + hf * 8 + j) * 168 + c] = f2bf(pv);
        }
    }
    __syncthreads();

    // out = P(16x160) x V(160x64): 5 K-chunks x 4 N-tiles
    v8f oa[4];
#pragma unroll
    for (int dt = 0; dt < 4; ++dt) oa[dt] = (v8f)0.0f;
    const unsigned int* prow = pU + (size_t)(wave * 16 + l15) * 84;
#pragma unroll
    for (int kc = 0; kc < 5; ++kc) {
        Frag ap;
#pragma unroll
        for (int j = 0; j < 8; ++j)
            ap.d[j] = prow[kc * 16 + ((j < 4) ? j : j + 4) + hf * 4];
#pragma unroll
        for (int dt = 0; dt < 4; ++dt) {
            Frag bv;
            const unsigned short* v16 = (const unsigned short*)vU;
#pragma unroll
            for (int s = 0; s < 16; ++s) bv.u[s] = v16[((s & 7) + hf * 8 + (s >> 3) * 16 + kc * 32) * 72 + dt * 16 + l15];
            oa[dt] = __builtin_amdgcn_wmma_f32_16x16x32_bf16(
                false, ap.v, false, bv.v, (short)0, oa[dt], false, false);
        }
    }
#pragma unroll
    for (int dt = 0; dt < 4; ++dt)
#pragma unroll
        for (int j = 0; j < 8; ++j) {
            int d = dt * 16 + l15, pr = p0 + hf * 8 + j;
            outb[((size_t)(b * HW + pr)) * 512 + h * 64 + d] = f2bf(oa[dt][j]);
        }
}

// ---------------------------------------------------------------------------
// Host orchestration
// ---------------------------------------------------------------------------
static void launch_gemm(const unsigned short* A, int lda, long long sAo, long long sAi,
                        const unsigned short* Bm, int ldb, long long sBo, long long sBi,
                        float* Cf, int ldc, long long sCo, long long sCi, int colMajor,
                        unsigned short* Cb, int ldcb, long long sCbo, long long sCbi,
                        const float* bias, float alpha,
                        int M, int N, int K, int batches, int innerCount, int nValid,
                        hipStream_t stream) {
    dim3 grid((N + 127) / 128, (M + 63) / 64, batches);
    cft_gemm_bf16<<<grid, 256, 0, stream>>>(A, lda, sAo, sAi, Bm, ldb, sBo, sBi,
                                            Cf, ldc, sCo, sCi, colMajor,
                                            Cb, ldcb, sCbo, sCbi,
                                            bias, alpha, M, N, K, innerCount, nValid);
}

extern "C" void kernel_launch(void* const* d_in, const int* in_sizes, int n_in,
                              void* d_out, int out_size, void* d_ws, size_t ws_size,
                              hipStream_t stream) {
    (void)in_sizes; (void)n_in; (void)out_size; (void)ws_size;

    const float* query     = (const float*)d_in[0];
    const float* key_value = (const float*)d_in[1];
    const float* dw_w      = (const float*)d_in[2];
    const float* dw_b      = (const float*)d_in[3];
    const float* pw_w      = (const float*)d_in[4];
    const float* pw_b      = (const float*)d_in[5];
    const float* ml1_w     = (const float*)d_in[6];
    const float* ml2_w     = (const float*)d_in[7];
    const float* align_w   = (const float*)d_in[8];
    const float* cf_w      = (const float*)d_in[9];
    const float* cf_b      = (const float*)d_in[10];
    const float* proj_w    = (const float*)d_in[11];
    const float* proj_b    = (const float*)d_in[12];

    const int Bn = 8, C = 512, H = 64, Wd = 64, HW = 4096, NC = 150, NCp = 160, heads = 8;
    const long long BHW = (long long)Bn * HW;

    float* out_t     = (float*)d_out;                          // [B,C,H,W]
    float* mask_map  = out_t + (size_t)Bn * C * HW;            // [B,NC,HW]
    float* attn_mean = mask_map + (size_t)Bn * NC * HW;        // [B,HW,NC]

    size_t off = 0;
    char* base = (char*)d_ws;
    auto take = [&](size_t bytes) -> char* {
        char* p = base + off;
        off = (off + bytes + 255) & ~(size_t)255;
        return p;
    };
    unsigned short* q_dw  = (unsigned short*)take((size_t)BHW * C * 2);      // dw out / attn out (reuse)
    unsigned short* q_bf  = (unsigned short*)take((size_t)BHW * C * 2);
    unsigned short* kv_in = (unsigned short*)take((size_t)BHW * C * 2);
    unsigned short* xk1   = (unsigned short*)take((size_t)BHW * C * 2);
    unsigned short* xk2   = (unsigned short*)take((size_t)BHW * C * 2);
    unsigned short* maskb = (unsigned short*)take((size_t)Bn * NCp * HW * 2);
    unsigned short* cff   = (unsigned short*)take((size_t)Bn * NCp * C * 2);
    float*          kvf   = (float*)take((size_t)Bn * NCp * 2 * C * 4);
    unsigned short* kTb   = (unsigned short*)take((size_t)Bn * heads * 64 * NCp * 2);
    unsigned short* vBb   = (unsigned short*)take((size_t)Bn * heads * NCp * 64 * 2);
    unsigned short* Wpw   = (unsigned short*)take((size_t)C * C * 2);
    unsigned short* Wml1  = (unsigned short*)take((size_t)C * 128 * 2);
    unsigned short* Wal   = (unsigned short*)take((size_t)C * 128 * 2);
    unsigned short* Wml2  = (unsigned short*)take((size_t)C * NCp * 2);
    unsigned short* Wcf   = (unsigned short*)take((size_t)C * 2 * C * 2);
    unsigned short* Wproj = (unsigned short*)take((size_t)C * C * 2);
    unsigned short* attn_out = q_dw;   // reuse (q_dw dead after pointwise GEMM)

    // zero the accumulation / padded buffers
    hipMemsetAsync(maskb, 0, (size_t)Bn * NCp * HW * 2, stream);
    hipMemsetAsync(attn_mean, 0, (size_t)Bn * HW * NC * 4, stream);

    // --- weight prep ---
    cft_wtrans<<<dim3((C * C + 255) / 256, 1), 256, 0, stream>>>(pw_w, Wpw, C, C, C, 0, 0);
    cft_wtrans<<<dim3((128 * 128 + 255) / 256, 4), 256, 0, stream>>>(ml1_w, Wml1, 128, 128, 128, 128 * 128, 128 * 128);
    cft_wtrans<<<dim3((128 * 128 + 255) / 256, 4), 256, 0, stream>>>(align_w, Wal, 128, 128, 128, 128 * 128, 128 * 128);
    cft_wtrans<<<dim3((C * NCp + 255) / 256, 1), 256, 0, stream>>>(ml2_w, Wml2, C, NC, NCp, 0, 0);
    cft_wtrans<<<dim3((C * 2 * C + 255) / 256, 1), 256, 0, stream>>>(cf_w, Wcf, C, 2 * C, 2 * C, 0, 0);
    cft_wtrans<<<dim3((C * C + 255) / 256, 1), 256, 0, stream>>>(proj_w, Wproj, C, C, C, 0, 0);

    // --- depthwise + layout conversions ---
    cft_dwconv3x3<<<dim3((unsigned)((BHW * C + 255) / 256)), 256, 0, stream>>>(query, dw_w, dw_b, q_dw, Bn, C, H, Wd);
    cft_nchw_to_pc<<<dim3((unsigned)((BHW * C + 255) / 256)), 256, 0, stream>>>(key_value, kv_in, Bn, C, HW);

    // --- pointwise q GEMM (fold attention scale 1/8) ---
    launch_gemm(q_dw, C, 0, 0, Wpw, C, 0, 0,
                nullptr, 0, 0, 0, 0, q_bf, C, 0, 0,
                pw_b, 0.125f, (int)BHW, C, C, 1, 1, C, stream);

    // --- grouped 1x1 convs (batched over b x group) ---
    launch_gemm(kv_in, C, (long long)HW * C, 128, Wml1, 128, 0, 128 * 128,
                nullptr, 0, 0, 0, 0, xk1, C, (long long)HW * C, 128,
                nullptr, 1.0f, HW, 128, 128, Bn * 4, 4, 128, stream);
    launch_gemm(kv_in, C, (long long)HW * C, 128, Wal, 128, 0, 128 * 128,
                nullptr, 0, 0, 0, 0, xk2, C, (long long)HW * C, 128,
                nullptr, 1.0f, HW, 128, 128, Bn * 4, 4, 128, stream);

    // --- ml2 -> mask_map (fp32, [B,NC,HW] => col-major epilogue) ---
    launch_gemm(xk1, C, (long long)HW * C, 0, Wml2, NCp, 0, 0,
                mask_map, HW, (long long)NC * HW, 0, 1, nullptr, 0, 0, 0,
                nullptr, 1.0f, HW, NCp, C, Bn, 1, NC, stream);

    // --- spatial softmax -> bf16 mask ---
    cft_softmax_hw<<<dim3(NC, Bn), 256, 0, stream>>>(mask_map, maskb, HW);

    // --- cf_feat = mask[160,4096] x xk2[4096,512] per batch ---
    launch_gemm(maskb, HW, (long long)NCp * HW, 0, xk2, C, (long long)HW * C, 0,
                nullptr, 0, 0, 0, 0, cff, C, (long long)NCp * C, 0,
                nullptr, 1.0f, NCp, C, HW, Bn, 1, C, stream);

    // --- kv = cf_feat[1280,512] x Wcf[512,1024] + cf_b ---
    launch_gemm(cff, C, 0, 0, Wcf, 2 * C, 0, 0,
                kvf, 2 * C, 0, 0, 0, nullptr, 0, 0, 0,
                cf_b, 1.0f, Bn * NCp, 2 * C, C, 1, 1, 2 * C, stream);

    // --- repack k/v into WMMA-friendly layouts ---
    cft_repack_kv<<<dim3((unsigned)(((size_t)Bn * NCp * 1024 + 255) / 256)), 256, 0, stream>>>(kvf, kTb, vBb, Bn);

    // --- fused attention (q*kT -> softmax -> attn_mean -> p*v) ---
    cft_attention<<<dim3(HW / 64, heads, Bn), 128, 0, stream>>>(q_bf, kTb, vBb, attn_out, attn_mean, HW);

    // --- proj GEMM -> NCHW fp32 output (col-major epilogue per batch) ---
    launch_gemm(attn_out, C, (long long)HW * C, 0, Wproj, C, 0, 0,
                out_t, HW, (long long)C * HW, 0, 1, nullptr, 0, 0, 0,
                proj_b, 1.0f, HW, C, C, Bn, 1, C, stream);
}
